// Model_79164837200451
// MI455X (gfx1250) — matrix operand order repaired and checked by reading, not compiled, not run on hardware
//
#include <hip/hip_runtime.h>
#include <hip/hip_bf16.h>

#define N_PTS 65536
#define C_CH  128
#define NS_NB 16
#define CS_CH 16
#define EPSV  1e-5f

typedef __attribute__((ext_vector_type(16))) __bf16 v16bf;
typedef __attribute__((ext_vector_type(8)))  __bf16 v8bf;
typedef __attribute__((ext_vector_type(8)))  float  v8f;

// ---------------- workspace layout (bytes) ----------------
#define WS_XB    ((size_t)0)                    // bf16 x            [N,128]   16 MB
#define WS_XQ    ((size_t)16777216)             // f32 xq            [N,128]   32 MB
#define WS_XK    ((size_t)50331648)             // f32 xk (pre-gather)         32 MB
#define WS_XV    ((size_t)83886080)             // f32 xv (pre-gather)         32 MB
#define WS_WCAT  ((size_t)117440512)            // bf16 [Wq;Wk;Wv]   [384,128] 96 KB
#define WS_BCAT  ((size_t)117538816)            // f32  [bq;bk;bv]   [384]
#define WS_WW1B  ((size_t)117540352)            // bf16 ww1          [16,128]
#define WS_PRM   ((size_t)117544448)            // f32 folded params [300]
// prm: [0..8]=pw1*s  [9..11]=pb1 folded  [12..139]=s1  [140..267]=h1
//      [268..283]=s2 [284..299]=h2

// ---------------- prep: fold BN, convert weights to bf16 ----------------
__global__ __launch_bounds__(256)
void k_prep(const float* __restrict__ Wq, const float* __restrict__ bq,
            const float* __restrict__ Wk, const float* __restrict__ bk,
            const float* __restrict__ Wv, const float* __restrict__ bv,
            const float* __restrict__ pw1, const float* __restrict__ pb1,
            const float* __restrict__ pbn_g, const float* __restrict__ pbn_b,
            const float* __restrict__ pbn_m, const float* __restrict__ pbn_v,
            const float* __restrict__ wbn1_g, const float* __restrict__ wbn1_b,
            const float* __restrict__ wbn1_m, const float* __restrict__ wbn1_v,
            const float* __restrict__ ww1,
            const float* __restrict__ wbn2_g, const float* __restrict__ wbn2_b,
            const float* __restrict__ wbn2_m, const float* __restrict__ wbn2_v,
            __bf16* __restrict__ Wcat, float* __restrict__ bcat,
            __bf16* __restrict__ ww1b, float* __restrict__ prm) {
  const int t = threadIdx.x;
  for (int e = t; e < C_CH * C_CH; e += 256) {
    Wcat[e]                 = (__bf16)Wq[e];
    Wcat[C_CH*C_CH + e]     = (__bf16)Wk[e];
    Wcat[2*C_CH*C_CH + e]   = (__bf16)Wv[e];
  }
  for (int e = t; e < C_CH; e += 256) {
    bcat[e] = bq[e]; bcat[C_CH + e] = bk[e]; bcat[2*C_CH + e] = bv[e];
  }
  for (int e = t; e < CS_CH * C_CH; e += 256) ww1b[e] = (__bf16)ww1[e];
  if (t < 3) {
    float s = pbn_g[t] * rsqrtf(pbn_v[t] + EPSV);
    prm[9 + t] = (pb1[t] - pbn_m[t]) * s + pbn_b[t];
    for (int e = 0; e < 3; ++e) prm[t*3 + e] = pw1[t*3 + e] * s;
  }
  if (t < C_CH) {
    float s = wbn1_g[t] * rsqrtf(wbn1_v[t] + EPSV);
    prm[12 + t]        = s;
    prm[140 + t]       = wbn1_b[t] - wbn1_m[t] * s;
  }
  if (t < CS_CH) {
    float s = wbn2_g[t] * rsqrtf(wbn2_v[t] + EPSV);
    prm[268 + t] = s;
    prm[284 + t] = wbn2_b[t] - wbn2_m[t] * s;
  }
}

// ---------------- convert x -> bf16 ----------------
__global__ __launch_bounds__(256)
void k_cvt_x(const float* __restrict__ x, __bf16* __restrict__ xb) {
  int e = (blockIdx.x * 256 + threadIdx.x) * 4;
  float4 v = *(const float4*)(x + e);
  xb[e+0] = (__bf16)v.x; xb[e+1] = (__bf16)v.y;
  xb[e+2] = (__bf16)v.z; xb[e+3] = (__bf16)v.w;
}

// ---------------- fused q/k/v projection GEMM (WMMA bf16) ----------------
// D[65536 x 384] = xb[65536 x 128] @ Wcat^T  (+ bcat)
// one wave per 16x16 output tile, K = 4 chunks of 32
__global__ __launch_bounds__(32)
void k_qkv(const __bf16* __restrict__ xb, const __bf16* __restrict__ Wcat,
           const float* __restrict__ bcat,
           float* __restrict__ xq, float* __restrict__ xkf, float* __restrict__ xvf) {
  const int tm   = blockIdx.x;        // 0..4095
  const int tn   = blockIdx.y;        // 0..23
  const int l    = threadIdx.x;
  const int lr   = l & 15;
  const int half = l >> 4;
  const int arow = tm * 16 + lr;      // A-matrix row held by this lane
  const int bcol = tn * 16 + lr;      // B column -> row of Wcat
  const int koff = half * 8;

  v8f acc = {};
#pragma unroll
  for (int kc = 0; kc < 4; ++kc) {
    // A frag: elems 0..15 -> K=kc*32+half*16+(0..15)
    const __bf16* ap = xb + (size_t)arow * C_CH + kc*32 + half*16;
    v8bf a0 = *(const v8bf*)(ap);
    v8bf a1 = *(const v8bf*)(ap + 8);
    v16bf a = __builtin_shufflevector(a0, a1, 0,1,2,3,4,5,6,7,8,9,10,11,12,13,14,15);
    // B frag: lanes 0-15 K=kc*32+(0..15), lanes 16-31 K=kc*32+16+(0..15)
    v16bf b = *(const v16bf*)(Wcat + (size_t)bcol * C_CH + kc*32 + half*16);
    acc = __builtin_amdgcn_wmma_f32_16x16x32_bf16(false, a, false, b,
                                                  (short)0, acc, false, false);
  }
  // D: col = lr (global tn*16+lr), rows = r + half*8
  const int gcol = tn * 16 + lr;
  const float bias = bcat[gcol];
  const int col = gcol & (C_CH - 1);
  float* outbuf = (tn < 8) ? xq : ((tn < 16) ? xkf : xvf);
#pragma unroll
  for (int r = 0; r < 8; ++r) {
    int row = tm * 16 + r + half * 8;
    outbuf[(size_t)row * C_CH + col] = acc[r] + bias;
  }
}

// ---------------- per-point attention (1 wave / point) ----------------
__global__ __launch_bounds__(32)
void k_attn(const float* __restrict__ p, const int* __restrict__ idx,
            const float* __restrict__ xq, const float* __restrict__ xkf,
            const float* __restrict__ xvf, const float* __restrict__ prm,
            const float* __restrict__ pw2, const float* __restrict__ pb2,
            const float* __restrict__ wb1, const float* __restrict__ ww2,
            const float* __restrict__ wb2, const __bf16* __restrict__ ww1b,
            float* __restrict__ out) {
  __shared__ int   idx_s[NS_NB];
  __shared__ float t_s[NS_NB][4];
  __shared__ float w1_s[NS_NB][CS_CH];
  __shared__ float soft_s[NS_NB][CS_CH];

  const int i    = blockIdx.x;
  const int l    = threadIdx.x;
  const int lr   = l & 15;
  const int half = l >> 4;
  const int koff = half * 8;

  const float* pw1s = prm;         // [3][3] (BN folded)
  const float* pb1s = prm + 9;
  const float* s1   = prm + 12;    // [128]
  const float* h1   = prm + 140;   // [128]
  const float* s2p  = prm + 268;   // [16]
  const float* h2p  = prm + 284;   // [16]

  // phase 0: relative positions + tiny folded conv -> t[j][0..2]
  const float px = p[i*3+0], py = p[i*3+1], pz = p[i*3+2];
  if (l < NS_NB) {
    int nj = idx[(size_t)i * NS_NB + l];
    idx_s[l] = nj;
    float rx = p[nj*3+0] - px, ry = p[nj*3+1] - py, rz = p[nj*3+2] - pz;
#pragma unroll
    for (int d = 0; d < 3; ++d) {
      float td = rx*pw1s[d*3+0] + ry*pw1s[d*3+1] + rz*pw1s[d*3+2] + pb1s[d];
      t_s[l][d] = fmaxf(td, 0.f);
    }
  }
  __syncthreads();

  const int   nrow = idx_s[lr];                 // neighbor for A row lr
  const float t0 = t_s[lr][0], t1 = t_s[lr][1], t2 = t_s[lr][2];

  // phase 1: w = relu(bn1(xk - xq + p_emb)) built directly in A-fragment
  // layout, then w @ ww1^T via 4 WMMAs (16x16x128)
  v8f acc = {};
#pragma unroll
  for (int kc = 0; kc < 4; ++kc) {
    v16bf a;
#pragma unroll
    for (int e8 = 0; e8 < 2; ++e8) {
      int c0 = kc*32 + half*16 + e8*8;
      float4 xka = *(const float4*)(xkf + (size_t)nrow * C_CH + c0);
      float4 xkb = *(const float4*)(xkf + (size_t)nrow * C_CH + c0 + 4);
      float4 xqa = *(const float4*)(xq  + (size_t)i    * C_CH + c0);
      float4 xqb = *(const float4*)(xq  + (size_t)i    * C_CH + c0 + 4);
      float xkv[8] = {xka.x,xka.y,xka.z,xka.w, xkb.x,xkb.y,xkb.z,xkb.w};
      float xqv[8] = {xqa.x,xqa.y,xqa.z,xqa.w, xqb.x,xqb.y,xqb.z,xqb.w};
#pragma unroll
      for (int u = 0; u < 8; ++u) {
        int c = c0 + u;
        float pe = t0*pw2[c*3+0] + t1*pw2[c*3+1] + t2*pw2[c*3+2] + pb2[c];
        float wv = xkv[u] - xqv[u] + pe;
        wv = fmaxf(wv * s1[c] + h1[c], 0.f);
        a[e8*8 + u] = (__bf16)wv;
      }
    }
    v16bf b = *(const v16bf*)(ww1b + (size_t)lr * C_CH + kc*32 + half*16);
    acc = __builtin_amdgcn_wmma_f32_16x16x32_bf16(false, a, false, b,
                                                  (short)0, acc, false, false);
  }

  // phase 2: +wb1, bn2, relu -> LDS (lane = col cs, rows r+half*8)
  {
    const float wb1c = wb1[lr], s2c = s2p[lr], h2c = h2p[lr];
#pragma unroll
    for (int r = 0; r < 8; ++r) {
      float v = acc[r] + wb1c;
      w1_s[r + half*8][lr] = fmaxf(v * s2c + h2c, 0.f);
    }
  }
  __syncthreads();

  // phase 3: 16x16 ww2 GEMM + softmax over the 16 neighbors (per col cs=lr)
  {
    float ww2r[CS_CH];
#pragma unroll
    for (int cs = 0; cs < CS_CH; ++cs) ww2r[cs] = ww2[lr*CS_CH + cs];
    const float wb2c = wb2[lr];
    float w2v[NS_NB];
    float mx = -3.0e38f;
#pragma unroll
    for (int j = 0; j < NS_NB; ++j) {
      float s = wb2c;
#pragma unroll
      for (int cs = 0; cs < CS_CH; ++cs) s += w1_s[j][cs] * ww2r[cs];
      w2v[j] = s;
      mx = fmaxf(mx, s);
    }
    float sum = 0.f;
#pragma unroll
    for (int j = 0; j < NS_NB; ++j) { w2v[j] = __expf(w2v[j] - mx); sum += w2v[j]; }
    float inv = 1.f / sum;
    if (half == 0) {
#pragma unroll
      for (int j = 0; j < NS_NB; ++j) soft_s[j][lr] = w2v[j] * inv;
    }
  }
  __syncthreads();

  // phase 4: out[c] = sum_j (xv[n_j][c] + p_emb[j][c]) * soft[j][c & 15]
  {
    const int cbase = l * 4;
    const int csb   = cbase & (CS_CH - 1);
    float pwc[4][3], pbc[4];
#pragma unroll
    for (int u = 0; u < 4; ++u) {
      pwc[u][0] = pw2[(cbase+u)*3+0];
      pwc[u][1] = pw2[(cbase+u)*3+1];
      pwc[u][2] = pw2[(cbase+u)*3+2];
      pbc[u]    = pb2[cbase+u];
    }
    float o0 = 0.f, o1 = 0.f, o2 = 0.f, o3 = 0.f;
#pragma unroll
    for (int j = 0; j < NS_NB; ++j) {
      int nj = idx_s[j];
      float4 xv4 = *(const float4*)(xvf + (size_t)nj * C_CH + cbase);
      float a0 = t_s[j][0], a1 = t_s[j][1], a2 = t_s[j][2];
      float sw0 = soft_s[j][csb+0], sw1 = soft_s[j][csb+1];
      float sw2 = soft_s[j][csb+2], sw3 = soft_s[j][csb+3];
      o0 += (xv4.x + a0*pwc[0][0] + a1*pwc[0][1] + a2*pwc[0][2] + pbc[0]) * sw0;
      o1 += (xv4.y + a0*pwc[1][0] + a1*pwc[1][1] + a2*pwc[1][2] + pbc[1]) * sw1;
      o2 += (xv4.z + a0*pwc[2][0] + a1*pwc[2][1] + a2*pwc[2][2] + pbc[2]) * sw2;
      o3 += (xv4.w + a0*pwc[3][0] + a1*pwc[3][1] + a2*pwc[3][2] + pbc[3]) * sw3;
    }
    *(float4*)(out + (size_t)i * C_CH + cbase) = make_float4(o0, o1, o2, o3);
  }
}

// ---------------- launch ----------------
extern "C" void kernel_launch(void* const* d_in, const int* in_sizes, int n_in,
                              void* d_out, int out_size, void* d_ws, size_t ws_size,
                              hipStream_t stream) {
  const float* p      = (const float*)d_in[0];
  const float* x      = (const float*)d_in[1];
  const int*   idx    = (const int*)  d_in[2];
  const float* Wq     = (const float*)d_in[3];
  const float* bq     = (const float*)d_in[4];
  const float* Wk     = (const float*)d_in[5];
  const float* bk     = (const float*)d_in[6];
  const float* Wv     = (const float*)d_in[7];
  const float* bv     = (const float*)d_in[8];
  const float* pw1    = (const float*)d_in[9];
  const float* pb1    = (const float*)d_in[10];
  const float* pbn_g  = (const float*)d_in[11];
  const float* pbn_b  = (const float*)d_in[12];
  const float* pbn_m  = (const float*)d_in[13];
  const float* pbn_v  = (const float*)d_in[14];
  const float* pw2    = (const float*)d_in[15];
  const float* pb2    = (const float*)d_in[16];
  const float* wbn1_g = (const float*)d_in[17];
  const float* wbn1_b = (const float*)d_in[18];
  const float* wbn1_m = (const float*)d_in[19];
  const float* wbn1_v = (const float*)d_in[20];
  const float* ww1    = (const float*)d_in[21];
  const float* wb1    = (const float*)d_in[22];
  const float* wbn2_g = (const float*)d_in[23];
  const float* wbn2_b = (const float*)d_in[24];
  const float* wbn2_m = (const float*)d_in[25];
  const float* wbn2_v = (const float*)d_in[26];
  const float* ww2    = (const float*)d_in[27];
  const float* wb2    = (const float*)d_in[28];

  char* ws = (char*)d_ws;
  __bf16* xb   = (__bf16*)(ws + WS_XB);
  float*  xq   = (float*) (ws + WS_XQ);
  float*  xkf  = (float*) (ws + WS_XK);
  float*  xvf  = (float*) (ws + WS_XV);
  __bf16* Wcat = (__bf16*)(ws + WS_WCAT);
  float*  bcat = (float*) (ws + WS_BCAT);
  __bf16* ww1b = (__bf16*)(ws + WS_WW1B);
  float*  prm  = (float*) (ws + WS_PRM);
  float*  out  = (float*)d_out;

  k_prep<<<1, 256, 0, stream>>>(Wq, bq, Wk, bk, Wv, bv,
                                pw1, pb1, pbn_g, pbn_b, pbn_m, pbn_v,
                                wbn1_g, wbn1_b, wbn1_m, wbn1_v, ww1,
                                wbn2_g, wbn2_b, wbn2_m, wbn2_v,
                                Wcat, bcat, ww1b, prm);

  k_cvt_x<<<(N_PTS * C_CH) / (256 * 4), 256, 0, stream>>>(x, xb);

  dim3 g1(N_PTS / 16, (3 * C_CH) / 16);
  k_qkv<<<g1, 32, 0, stream>>>(xb, Wcat, bcat, xq, xkf, xvf);

  k_attn<<<N_PTS, 32, 0, stream>>>(p, idx, xq, xkf, xvf, prm,
                                   pw2, pb2, wb1, ww2, wb2, ww1b, out);
}
